// SpatialTransformer_19653770346841
// MI455X (gfx1250) — matrix operand order repaired and checked by reading, not compiled, not run on hardware
//
#include <hip/hip_runtime.h>

// ---------------------------------------------------------------------------
// Types for CDNA5 WMMA (wave32, 16x16x32 bf16 -> f32)
// ---------------------------------------------------------------------------
typedef __attribute__((ext_vector_type(16))) __bf16 bf16x16;
typedef __attribute__((ext_vector_type(8)))  __bf16 bf16x8;
typedef __attribute__((ext_vector_type(8)))  float  f32x8;

static __device__ inline f32x8 wmma_bf16(bf16x16 a, bf16x16 b, f32x8 c) {
    // (neg_a, A, neg_b, B, c_mod, C, reuse_a, reuse_b)
    return __builtin_amdgcn_wmma_f32_16x16x32_bf16(false, a, false, b, (short)0, c, false, false);
}

static __device__ inline bf16x8 ld8(const __bf16* p) {
    return *reinterpret_cast<const bf16x8*>(p);
}
static __device__ inline void st8(__bf16* p, bf16x8 v) {
    *reinterpret_cast<bf16x8*>(p) = v;
}

// CDNA5 async global->LDS copy (CDNA5 ISA, VGLOBAL op 98).
// 16 bytes per active lane; LDS dest address = low 32 bits of generic pointer
// (flat-LDS truncation rule, ISA §10.2). Tracked by ASYNCcnt.
static __device__ inline void async_ld128(void* lds, const void* gaddr) {
    unsigned int l = (unsigned int)(unsigned long long)lds;
    asm volatile("global_load_async_to_lds_b128 %0, %1, off"
                 :: "v"(l), "v"(gaddr) : "memory");
}
static __device__ inline void wait_async() {
    asm volatile("s_wait_asynccnt 0x0" ::: "memory");
}

// A-fragment (16x32, M x K). lane m = lane&15, bk = (lane>>4)*8.
// element j in [0,15] -> K = 2*bk + j (contiguous, matches mkB)
static __device__ inline bf16x16 mkA(const __bf16* row, int bk) {
    bf16x8 lo = ld8(row + 2 * bk);
    bf16x8 hi = ld8(row + 2 * bk + 8);
    bf16x16 a;
#pragma unroll
    for (int j = 0; j < 8; ++j) { a[j] = lo[j]; a[8 + j] = hi[j]; }
    return a;
}

// B-fragment (32x16, K x N). lane n = lane&15 holds K = (lane>=16?16:0)+j, j=0..15
static __device__ inline bf16x16 mkB(const __bf16* p) {
    bf16x8 lo = ld8(p);
    bf16x8 hi = ld8(p + 8);
    bf16x16 b;
#pragma unroll
    for (int j = 0; j < 8; ++j) { b[j] = lo[j]; b[8 + j] = hi[j]; }
    return b;
}

// Problem constants
#define BATCH 2
#define H 256
#define W 256
#define HW 65536
#define CIN 64
#define DH 256         // per-branch channels (head*64)
#define QKV 768        // 3*DH packed
#define NWIN 256       // 16x16 windows
#define HEADS 4
#define TOK 256        // tokens per window

// ---------------------------------------------------------------------------
// Weight prep: fp32 -> bf16, WMMA-friendly layouts
// ---------------------------------------------------------------------------
__global__ __launch_bounds__(256) void k_prep(
    const float* __restrict__ wq1, const float* __restrict__ wk1, const float* __restrict__ wv1,
    const float* __restrict__ bq1, const float* __restrict__ bk1, const float* __restrict__ bv1,
    const float* __restrict__ wq2, const float* __restrict__ wk2, const float* __restrict__ wv2,
    const float* __restrict__ wp,
    __bf16* __restrict__ w1b, __bf16* __restrict__ w2b, __bf16* __restrict__ wpb,
    float* __restrict__ b1c)
{
    const int n_w1 = QKV * CIN;          // 49152
    const int n_w2 = 3 * 9 * DH * DH;    // 1769472
    const int n_wp = 64 * DH;            // 16384
    const int n_b1 = QKV;                // 768
    int tid = blockIdx.x * 256 + threadIdx.x;
    if (tid < n_w1) {
        int oc = tid >> 6, ic = tid & 63;
        float v;
        if      (oc < 256) v = wq1[oc * 64 + ic];
        else if (oc < 512) v = wk1[(oc - 256) * 64 + ic];
        else               v = wv1[(oc - 512) * 64 + ic];
        w1b[tid] = (__bf16)v;
        return;
    }
    tid -= n_w1;
    if (tid < n_w2) {
        int conv = tid / (9 * DH * DH);
        int rem  = tid - conv * (9 * DH * DH);
        int tap  = rem >> 16;            // /65536
        int oi   = rem & 65535;
        int oc = oi >> 8, ic = oi & 255;
        const float* src = (conv == 0) ? wq2 : (conv == 1) ? wk2 : wv2;
        w2b[(size_t)conv * (9 * DH * DH) + tap * (DH * DH) + oc * DH + ic] =
            (__bf16)src[oc * (DH * 9) + ic * 9 + tap];
        return;
    }
    tid -= n_w2;
    if (tid < n_wp) { wpb[tid] = (__bf16)wp[tid]; return; }
    tid -= n_wp;
    if (tid < n_b1) {
        float v;
        if      (tid < 256) v = bq1[tid];
        else if (tid < 512) v = bk1[tid - 256];
        else                v = bv1[tid - 512];
        b1c[tid] = v;
    }
}

// ---------------------------------------------------------------------------
// Pack x: NCHW f32 -> NHWC bf16
// ---------------------------------------------------------------------------
__global__ __launch_bounds__(256) void k_pack(const float* __restrict__ x,
                                              __bf16* __restrict__ xp)
{
    int tid = blockIdx.x * 256 + threadIdx.x;     // over BATCH*HW*CIN = 8388608
    int c = tid & 63;
    int bp = tid >> 6;
    int p = bp & (HW - 1);
    int b = bp >> 16;
    xp[tid] = (__bf16)x[((size_t)(b * CIN + c)) * HW + p];
}

// ---------------------------------------------------------------------------
// Fused q|k|v 1x1 conv: t[pix][0..767] = W1 * x[pix] + b1   (bf16 out)
// grid (2048, 6), block 256 (8 waves). wave = 16 oc, 4 N-tiles of 16 pixels.
// ---------------------------------------------------------------------------
__global__ __launch_bounds__(256) void k_conv1(const __bf16* __restrict__ xp,
                                               const __bf16* __restrict__ w1b,
                                               const float* __restrict__ b1c,
                                               __bf16* __restrict__ t)
{
    const int lane = threadIdx.x & 31;
    const int wave = threadIdx.x >> 5;
    const int oc0  = blockIdx.y * 128 + wave * 16;
    const int pg   = blockIdx.x;                 // 64 pixels per group
    const int nl   = lane & 15;
    const int half = lane >> 4;
    const int bk   = half * 8;

    const __bf16* wrow = w1b + (size_t)(oc0 + nl) * 64;
    bf16x16 a0 = mkA(wrow, bk);
    bf16x16 a1 = mkA(wrow + 32, bk);

#pragma unroll
    for (int nt = 0; nt < 4; ++nt) {
        int pix = pg * 64 + nt * 16 + nl;
        const __bf16* xr = xp + (size_t)pix * 64;
        bf16x16 b0 = mkB(xr + half * 16);
        bf16x16 b1 = mkB(xr + 32 + half * 16);
        f32x8 acc = 0.0f;
        acc = wmma_bf16(a0, b0, acc);
        acc = wmma_bf16(a1, b1, acc);
#pragma unroll
        for (int r = 0; r < 8; ++r) {
            int oc = oc0 + r + half * 8;
            t[(size_t)pix * QKV + oc] = (__bf16)(acc[r] + b1c[oc]);
        }
    }
}

// ---------------------------------------------------------------------------
// Dilated 3x3 conv (K = 9*256) + bias + optional per-(pixel,head) L2 norm.
// Double-buffered async global->LDS staging: tap t+1 streams into LDS via
// ASYNCcnt while tap t runs 8 WMMAs per wave. grid 8192, block 512 (16 waves).
// ---------------------------------------------------------------------------
__global__ __launch_bounds__(512) void k_conv3(const __bf16* __restrict__ t, int coff,
                                               const __bf16* __restrict__ w2,
                                               const float* __restrict__ b2,
                                               int dil, int do_norm,
                                               __bf16* __restrict__ outw)
{
    __shared__ __bf16 Bs[2][16][264];   // double-buffered [pixel][ic] tile
    __shared__ float  Cs[256][17];      // [oc][pixel] accum for norm
    __shared__ float  Sc[4][16];        // [head][pixel] scale

    const int lane = threadIdx.x & 31;
    const int wave = threadIdx.x >> 5;
    const int oc0  = wave * 16;
    const int rt = blockIdx.x;
    const int xt = rt & 15, y = (rt >> 4) & 255, b = rt >> 12;
    const int nl = lane & 15, half = lane >> 4, bk = half * 8;

    // this thread's fixed slot in the 16px x 256ic tile (8 bf16 = 16B)
    const int flat = threadIdx.x * 8;
    const int sn = flat >> 8, ic = flat & 255;

    // stage tile for `tap` into buffer `buf` (async for in-bounds lanes)
    auto stage = [&](int tap, int buf) {
        int ty = tap / 3, tx = tap - ty * 3;
        int sy = y + (ty - 1) * dil;
        int sx = xt * 16 + sn + (tx - 1) * dil;
        __bf16* dst = &Bs[buf][sn][ic];
        if (sy >= 0 && sy < H && sx >= 0 && sx < W) {
            const __bf16* src = t + ((size_t)(b * HW + sy * W + sx)) * QKV + coff + ic;
            async_ld128(dst, src);
            __builtin_prefetch(src + QKV, 0, 1);   // global_prefetch next row
        } else {
            st8(dst, (bf16x8)(__bf16)0.0f);        // zero padding via DS store
        }
    };

    f32x8 acc = 0.0f;

    stage(0, 0);
    wait_async();
    __syncthreads();

    for (int tap = 0; tap < 9; ++tap) {
        const int buf = tap & 1;
        if (tap < 8) stage(tap + 1, buf ^ 1);      // overlap with WMMAs below

        const __bf16* wrow = w2 + (size_t)tap * (DH * DH) + (size_t)(oc0 + nl) * DH;
#pragma unroll
        for (int kk = 0; kk < 8; ++kk) {
            bf16x16 a  = mkA(wrow + kk * 32, bk);
            bf16x16 bf = mkB(&Bs[buf][nl][kk * 32 + half * 16]);
            acc = wmma_bf16(a, bf, acc);
        }
        wait_async();      // own async writes to next buffer landed in LDS
        __syncthreads();   // all waves done staging + done reading `buf`
    }

#pragma unroll
    for (int r = 0; r < 8; ++r) {
        int oc = oc0 + r + half * 8;
        Cs[oc][nl] = acc[r] + b2[oc];
    }
    __syncthreads();
    if (threadIdx.x < 64) {
        int px = threadIdx.x & 15, hd = threadIdx.x >> 4;
        float s = 1.0f;
        if (do_norm) {
            float ss = 0.0f;
            for (int c = 0; c < 64; ++c) { float v = Cs[hd * 64 + c][px]; ss += v * v; }
            float nrm = sqrtf(ss);
            nrm = fmaxf(nrm, 1e-12f);
            s = 1.0f / nrm;
        }
        Sc[hd][px] = s;
    }
    __syncthreads();

    const int win = (y >> 4) * 16 + xt;
    const int token = (y & 15) * 16 + nl;
#pragma unroll
    for (int r = 0; r < 8; ++r) {
        int oc = oc0 + r + half * 8;
        int hd = oc >> 6, c = oc & 63;
        float v = Cs[oc][nl] * Sc[hd][nl];
        outw[((((size_t)(b * NWIN + win)) * HEADS + hd) * TOK + token) * 64 + c] = (__bf16)v;
    }
}

// ---------------------------------------------------------------------------
// Windowed attention: per block one (b, win, head); 256x64 q,k,v.
// S = softmax(0.25 * q k^T), O = S v. block 512 (16 waves, 16 q-rows each).
// ---------------------------------------------------------------------------
__global__ __launch_bounds__(512) void k_attn(const __bf16* __restrict__ qw,
                                              const __bf16* __restrict__ kw,
                                              const __bf16* __restrict__ vw,
                                              __bf16* __restrict__ ow)
{
    __shared__ __bf16 Ks[256][72];        // [token][c]
    __shared__ __bf16 Vt[64][264];        // [c][token] (transposed V)
    __shared__ __bf16 Ps[16][16][32];     // per-wave P tile staging (16x32)

    const int tid = threadIdx.x;
    const int lane = tid & 31, wave = tid >> 5;
    const int bx = blockIdx.x;
    const int head = bx & 3, win = (bx >> 2) & 255, b = bx >> 10;
    const size_t base = ((((size_t)b * NWIN + win) * HEADS + head) * TOK) * 64;
    const __bf16* qb = qw + base;
    const __bf16* kb = kw + base;
    const __bf16* vb = vw + base;

    // stage K via async global->LDS, V transposed via DS scatter
#pragma unroll
    for (int i = 0; i < 4; ++i) {
        int flat = (i * 512 + tid) * 8;
        int tok = flat >> 6, c = flat & 63;
        async_ld128(&Ks[tok][c], kb + flat);
    }
#pragma unroll
    for (int i = 0; i < 32; ++i) {
        int flat = i * 512 + tid;
        int tok = flat >> 6, c = flat & 63;
        Vt[c][tok] = vb[flat];
    }
    wait_async();
    __syncthreads();

    const int nl = lane & 15, half = lane >> 4, bk = half * 8;
    const int m0 = wave * 16;
    const __bf16* qrow = qb + (size_t)(m0 + nl) * 64;
    bf16x16 aq0 = mkA(qrow, bk);
    bf16x16 aq1 = mkA(qrow + 32, bk);

    f32x8 s[16];
#pragma unroll
    for (int ct = 0; ct < 16; ++ct) {
        bf16x16 b0 = mkB(&Ks[ct * 16 + nl][half * 16]);
        bf16x16 b1 = mkB(&Ks[ct * 16 + nl][32 + half * 16]);
        f32x8 z = 0.0f;
        z = wmma_bf16(aq0, b0, z);
        s[ct] = wmma_bf16(aq1, b1, z);
    }

    // in-register softmax; rescale = 0.25 folded into exp.
#pragma unroll
    for (int r = 0; r < 8; ++r) {
        float mx = s[0][r];
#pragma unroll
        for (int ct = 1; ct < 16; ++ct) mx = fmaxf(mx, s[ct][r]);
        mx = fmaxf(mx, __shfl_xor(mx, 1, 32));
        mx = fmaxf(mx, __shfl_xor(mx, 2, 32));
        mx = fmaxf(mx, __shfl_xor(mx, 4, 32));
        mx = fmaxf(mx, __shfl_xor(mx, 8, 32));
        float sum = 0.0f;
#pragma unroll
        for (int ct = 0; ct < 16; ++ct) {
            float e = __expf(0.25f * (s[ct][r] - mx));
            s[ct][r] = e;
            sum += e;
        }
        sum += __shfl_xor(sum, 1, 32);
        sum += __shfl_xor(sum, 2, 32);
        sum += __shfl_xor(sum, 4, 32);
        sum += __shfl_xor(sum, 8, 32);
        float inv = 1.0f / sum;
#pragma unroll
        for (int ct = 0; ct < 16; ++ct) s[ct][r] *= inv;
    }

    // O = P * V : k-chunks of 32 tokens; P -> A-fragment via wave-private LDS
    f32x8 o0 = 0.0f, o1 = 0.0f, o2 = 0.0f, o3 = 0.0f;
#pragma unroll
    for (int kc = 0; kc < 8; ++kc) {
#pragma unroll
        for (int r = 0; r < 8; ++r) {
            int row = r + half * 8;
            Ps[wave][row][nl]      = (__bf16)s[2 * kc][r];
            Ps[wave][row][16 + nl] = (__bf16)s[2 * kc + 1][r];
        }
        // wave-private region: DS ops are in-order within a wave, no barrier
        bf16x16 ap = mkA(&Ps[wave][nl][0], bk);
        o0 = wmma_bf16(ap, mkB(&Vt[ 0 + nl][kc * 32 + half * 16]), o0);
        o1 = wmma_bf16(ap, mkB(&Vt[16 + nl][kc * 32 + half * 16]), o1);
        o2 = wmma_bf16(ap, mkB(&Vt[32 + nl][kc * 32 + half * 16]), o2);
        o3 = wmma_bf16(ap, mkB(&Vt[48 + nl][kc * 32 + half * 16]), o3);
    }

#pragma unroll
    for (int r = 0; r < 8; ++r) {
        int token = m0 + r + half * 8;
        ow[base + (size_t)token * 64 +  0 + nl] = (__bf16)o0[r];
        ow[base + (size_t)token * 64 + 16 + nl] = (__bf16)o1[r];
        ow[base + (size_t)token * 64 + 32 + nl] = (__bf16)o2[r];
        ow[base + (size_t)token * 64 + 48 + nl] = (__bf16)o3[r];
    }
}

// ---------------------------------------------------------------------------
// Projection 1x1 (256 -> 64) reading window layout, writing NCHW f32 output.
// ---------------------------------------------------------------------------
__global__ __launch_bounds__(128) void k_proj(const __bf16* __restrict__ ow,
                                              const __bf16* __restrict__ wpb,
                                              const float* __restrict__ bp,
                                              float* __restrict__ out)
{
    const int lane = threadIdx.x & 31;
    const int wave = threadIdx.x >> 5;
    const int oc0 = wave * 16;
    const int rt = blockIdx.x;
    const int xt = rt & 15, y = (rt >> 4) & 255, b = rt >> 12;
    const int nl = lane & 15, half = lane >> 4, bk = half * 8;

    const int win = (y >> 4) * 16 + xt;
    const int token = (y & 15) * 16 + nl;
    const size_t pbase = ((size_t)(b * NWIN + win)) * (HEADS * TOK * 64);

    const __bf16* wrow = wpb + (size_t)(oc0 + nl) * 256;
    f32x8 acc = 0.0f;
#pragma unroll
    for (int kk = 0; kk < 8; ++kk) {
        int k0 = kk * 32;
        bf16x16 a = mkA(wrow + k0, bk);
        int kh = k0 + half * 16;
        int hd = kh >> 6, c = kh & 63;
        bf16x16 bf = mkB(ow + pbase + ((size_t)hd * TOK + token) * 64 + c);
        acc = wmma_bf16(a, bf, acc);
    }
#pragma unroll
    for (int r = 0; r < 8; ++r) {
        int oc = oc0 + r + half * 8;
        out[((size_t)(b * 64 + oc)) * HW + y * W + (xt * 16 + nl)] = acc[r] + bp[oc];
    }
}

// ---------------------------------------------------------------------------
// Host launcher
// ---------------------------------------------------------------------------
extern "C" void kernel_launch(void* const* d_in, const int* in_sizes, int n_in,
                              void* d_out, int out_size, void* d_ws, size_t ws_size,
                              hipStream_t stream)
{
    const float* x   = (const float*)d_in[0];
    const float* wq1 = (const float*)d_in[1];  const float* bq1 = (const float*)d_in[2];
    const float* wq2 = (const float*)d_in[3];  const float* bq2 = (const float*)d_in[4];
    const float* wk1 = (const float*)d_in[5];  const float* bk1 = (const float*)d_in[6];
    const float* wk2 = (const float*)d_in[7];  const float* bk2 = (const float*)d_in[8];
    const float* wv1 = (const float*)d_in[9];  const float* bv1 = (const float*)d_in[10];
    const float* wv2 = (const float*)d_in[11]; const float* bv2 = (const float*)d_in[12];
    const float* wp  = (const float*)d_in[13]; const float* bp  = (const float*)d_in[14];
    float* out = (float*)d_out;

    char* ws = (char*)d_ws;
    size_t off = 0;
    auto alloc = [&](size_t bytes) { size_t r = off; off = (off + bytes + 255) & ~(size_t)255; return r; };

    const size_t NPIX = (size_t)BATCH * HW;
    __bf16* xp  = (__bf16*)(ws + alloc(NPIX * CIN * 2));          //  16.8 MB
    __bf16* t   = (__bf16*)(ws + alloc(NPIX * QKV * 2));          // 201.3 MB
    __bf16* qwb = (__bf16*)(ws + alloc(NPIX * DH * 2));           //  67.1 MB
    __bf16* kwb = (__bf16*)(ws + alloc(NPIX * DH * 2));
    __bf16* vwb = (__bf16*)(ws + alloc(NPIX * DH * 2));
    __bf16* owb = (__bf16*)(ws + alloc(NPIX * DH * 2));
    __bf16* w1b = (__bf16*)(ws + alloc((size_t)QKV * CIN * 2));
    __bf16* w2b = (__bf16*)(ws + alloc((size_t)3 * 9 * DH * DH * 2));
    __bf16* wpb = (__bf16*)(ws + alloc((size_t)64 * DH * 2));
    float*  b1c = (float*)(ws + alloc((size_t)QKV * 4));
    (void)ws_size; (void)in_sizes; (void)n_in; (void)out_size;

    // 1) weight prep
    {
        int total = QKV * CIN + 3 * 9 * DH * DH + 64 * DH + QKV;
        k_prep<<<(total + 255) / 256, 256, 0, stream>>>(
            wq1, wk1, wv1, bq1, bk1, bv1, wq2, wk2, wv2, wp, w1b, w2b, wpb, b1c);
    }
    // 2) pack input
    k_pack<<<(int)((NPIX * CIN) / 256), 256, 0, stream>>>(x, xp);
    // 3) fused q|k|v 1x1 conv
    k_conv1<<<dim3(2048, 6), 256, 0, stream>>>(xp, w1b, b1c, t);
    // 4) dilated 3x3 convs -> window layout (+ L2 norm for q,k)
    const size_t wstride = (size_t)9 * DH * DH;
    k_conv3<<<8192, 512, 0, stream>>>(t,   0, w2b,               bq2, 3, 1, qwb);
    k_conv3<<<8192, 512, 0, stream>>>(t, 256, w2b + wstride,     bk2, 2, 1, kwb);
    k_conv3<<<8192, 512, 0, stream>>>(t, 512, w2b + 2 * wstride, bv2, 1, 0, vwb);
    // 5) windowed attention
    k_attn<<<BATCH * NWIN * HEADS, 512, 0, stream>>>(qwb, kwb, vwb, owb);
    // 6) projection
    k_proj<<<8192, 128, 0, stream>>>(owb, wpb, bp, out);
}
